// RBFNet_13597866459168
// MI455X (gfx1250) — compile-verified, matrix operand order checked by reading, not run on hardware
//
#include <hip/hip_runtime.h>

// ---- RBFNet fused kernel for MI455X (gfx1250, wave32, WMMA bf16) ----
// out = exp(-||x-loc||/scale) @ w^T  with B=8192, C=4096, F=1024, OUT=10
// Big GEMM (x @ loc^T) done as bf16x3 split (hi*hi + hi*lo + lo*hi) on
// v_wmma_f32_16x16x32_bf16; distances + RBF fused in-register; second
// GEMM (phi @ w^T) also via WMMA with OUT padded to 16.

typedef __attribute__((ext_vector_type(16))) __bf16 v16bf;
typedef __attribute__((ext_vector_type(8)))  __bf16 v8bf;
typedef __attribute__((ext_vector_type(4)))  __bf16 v4bf;
typedef __attribute__((ext_vector_type(8)))  float  v8f;

#define B_   8192
#define C_   4096
#define F_   1024
#define OUT_ 10

static __device__ __forceinline__ unsigned short f2bf_bits(float f) {
  unsigned u = __builtin_bit_cast(unsigned, f);
  u += 0x7FFFu + ((u >> 16) & 1u);          // round-to-nearest-even
  return (unsigned short)(u >> 16);
}
static __device__ __forceinline__ __bf16 f2bf(float f) {
  unsigned short b = f2bf_bits(f);
  return __builtin_bit_cast(__bf16, b);
}
static __device__ __forceinline__ float bfbits2f(unsigned short b) {
  unsigned u = ((unsigned)b) << 16;
  return __builtin_bit_cast(float, u);
}

// Load one 16x32 bf16 A/B fragment from row-major LDS.
// Fragment layout (ISA 7.12.2, 16-bit A 16x32): lane L holds row (L&15);
// elements 0..7 -> K = 8*(L>>4)+0..7, elements 8..15 -> K = 16+8*(L>>4)+0..7.
// B (32x16) is the mirror with N in place of M, so the same loader works for
// row-major loc / w / phi tiles.
static __device__ __forceinline__ v16bf load_frag(const __bf16* base, int strideH,
                                                  int row, int k0, int lane) {
  const __bf16* p = base + row * strideH + k0 + ((lane >> 4) << 3);
  v8bf a = *(const v8bf*)(p);
  v8bf b = *(const v8bf*)(p + 16);
  return __builtin_shufflevector(a, b, 0,1,2,3,4,5,6,7,8,9,10,11,12,13,14,15);
}

// ---- row sum-of-squares: one wave (32 lanes) per row of F_=1024 floats ----
__global__ __launch_bounds__(256) void rowsumsq_kernel(const float* __restrict__ src,
                                                       float* __restrict__ dst) {
  const int lane = threadIdx.x & 31;
  const int wid  = threadIdx.x >> 5;
  const int row  = blockIdx.x * 8 + wid;
  const float4* p = (const float4*)(src + (size_t)row * F_);
  float s = 0.f;
#pragma unroll
  for (int j = 0; j < 8; ++j) {
    float4 v = p[lane + 32 * j];
    s += v.x * v.x + v.y * v.y + v.z * v.z + v.w * v.w;
  }
#pragma unroll
  for (int off = 16; off > 0; off >>= 1) s += __shfl_xor(s, off, 32);
  if (lane == 0) dst[row] = s;
}

// ---- fused: GEMM(bf16x3) -> dist -> phi -> GEMM(phi @ w^T) ----
__global__ __launch_bounds__(256) void rbf_fused_kernel(
    const float* __restrict__ x, const float* __restrict__ loc,
    const float* __restrict__ scale, const float* __restrict__ w,
    const float* __restrict__ x2g, const float* __restrict__ c2g,
    float* __restrict__ part) {
  // LDS: region0 (40960B) = A_hi/A_lo/B_hi/B_lo 128x32 bf16 tiles (stride 40),
  //      reused as phi 128x128 bf16 (stride 136) after the K loop.
  //      region1 = w tile 16x128 bf16 (stride 136, rows >=10 zero).
  //      region2 = x2/c2/scale vectors (128 floats each).
  __shared__ __align__(16) char smem[46848];
  __bf16* Ahi  = (__bf16*)smem;           // 128*40 halfs
  __bf16* Alo  = Ahi + 5120;
  __bf16* Bhi  = Ahi + 10240;
  __bf16* Blo  = Ahi + 15360;
  __bf16* phiL = (__bf16*)smem;           // 128*136 halfs (reuse)
  __bf16* wL   = (__bf16*)(smem + 40960); // 16*136 halfs
  float*  vx2  = (float*)(smem + 45312);
  float*  vc2  = vx2 + 128;
  float*  vsc  = vc2 + 128;

  const int t    = threadIdx.x;
  const int lane = t & 31;
  const int wid  = t >> 5;          // 8 waves
  const int half = lane >> 4;
  const int ln   = lane & 15;
  const int rb   = blockIdx.x * 128;
  const int wm   = wid & 3;         // wave grid 4(M) x 2(N) over 128x128 tile
  const int wn   = wid >> 2;

  if (t < 128) vx2[t] = x2g[rb + t];

  v8f oacc = {};                    // wave's 16x16 out accumulator

  for (int cc = 0; cc < 4; ++cc) {
    const int cb = blockIdx.y * 512 + cc * 128;
    __syncthreads();                // protect wL/phi reads of previous iter
    if (t < 128) { vc2[t] = c2g[cb + t]; vsc[t] = scale[cb + t]; }
    for (int idx = t; idx < 16 * 128; idx += 256) {
      int o = idx >> 7, c = idx & 127;
      float v = (o < OUT_) ? w[o * C_ + cb + c] : 0.0f;
      wL[o * 136 + c] = f2bf(v);
    }

    v8f acc[2][4] = {};

    const int rA  = t >> 1;
    const int c16 = (t & 1) * 16;

    for (int kc = 0; kc < F_ / 32; ++kc) {
      const int k0 = kc * 32;
      __syncthreads();
      // stage A (x) and B (loc) 128x32 chunks, hi/lo bf16 split
      const float4* gA = (const float4*)(x   + (size_t)(rb + rA) * F_ + k0 + c16);
      const float4* gB = (const float4*)(loc + (size_t)(cb + rA) * F_ + k0 + c16);
#pragma unroll
      for (int q = 0; q < 4; ++q) {
        float4 va = gA[q];
        float4 vb = gB[q];
        float fa[4] = {va.x, va.y, va.z, va.w};
        float fb[4] = {vb.x, vb.y, vb.z, vb.w};
        v4bf ah, al, bh, bl;
#pragma unroll
        for (int e = 0; e < 4; ++e) {
          unsigned short hb = f2bf_bits(fa[e]);
          ah[e] = __builtin_bit_cast(__bf16, hb);
          al[e] = f2bf(fa[e] - bfbits2f(hb));
          unsigned short hc = f2bf_bits(fb[e]);
          bh[e] = __builtin_bit_cast(__bf16, hc);
          bl[e] = f2bf(fb[e] - bfbits2f(hc));
        }
        const int off = rA * 40 + c16 + q * 4;
        *(v4bf*)(Ahi + off) = ah;
        *(v4bf*)(Alo + off) = al;
        *(v4bf*)(Bhi + off) = bh;
        *(v4bf*)(Blo + off) = bl;
      }
      __syncthreads();
      // wave computes its 32x64 sub-tile: 2x4 wmma tiles, bf16x3 per tile
      v16bf aH[2], aL[2];
#pragma unroll
      for (int mi = 0; mi < 2; ++mi) {
        int row = wm * 32 + mi * 16 + ln;
        aH[mi] = load_frag(Ahi, 40, row, 0, lane);
        aL[mi] = load_frag(Alo, 40, row, 0, lane);
      }
#pragma unroll
      for (int ni = 0; ni < 4; ++ni) {
        int row = wn * 64 + ni * 16 + ln;
        v16bf bH = load_frag(Bhi, 40, row, 0, lane);
        v16bf bL = load_frag(Blo, 40, row, 0, lane);
#pragma unroll
        for (int mi = 0; mi < 2; ++mi) {
          acc[mi][ni] = __builtin_amdgcn_wmma_f32_16x16x32_bf16(
              false, aH[mi], false, bH, (short)0, acc[mi][ni], false, false);
          acc[mi][ni] = __builtin_amdgcn_wmma_f32_16x16x32_bf16(
              false, aH[mi], false, bL, (short)0, acc[mi][ni], false, false);
          acc[mi][ni] = __builtin_amdgcn_wmma_f32_16x16x32_bf16(
              false, aL[mi], false, bH, (short)0, acc[mi][ni], false, false);
        }
      }
    }

    // epilogue: sq -> dist -> phi, write bf16 phi tile to LDS (reuses A/B)
    __syncthreads();
#pragma unroll
    for (int mi = 0; mi < 2; ++mi) {
#pragma unroll
      for (int ni = 0; ni < 4; ++ni) {
#pragma unroll
        for (int r = 0; r < 8; ++r) {
          // C/D layout: VGPR r, lane -> M = r + 8*(lane>>4), N = lane&15
          int M = wm * 32 + mi * 16 + r + 8 * half;
          int N = wn * 64 + ni * 16 + ln;
          float dot  = acc[mi][ni][r];
          float sq   = vx2[M] + vc2[N] - 2.0f * dot;
          float dist = __builtin_sqrtf(fmaxf(sq, 0.0f));
          float phi  = __expf(-dist / vsc[N]);
          phiL[M * 136 + N] = f2bf(phi);
        }
      }
    }
    __syncthreads();
    // second GEMM: wave wid owns rows [16*wid, 16*wid+16); K = 128 this chunk
#pragma unroll
    for (int kc = 0; kc < 4; ++kc) {
      v16bf aF = load_frag(phiL, 136, wid * 16 + ln, kc * 32, lane);
      v16bf bF = load_frag(wL,   136, ln,            kc * 32, lane);
      oacc = __builtin_amdgcn_wmma_f32_16x16x32_bf16(
          false, aF, false, bF, (short)0, oacc, false, false);
    }
  }

  // deterministic partials: one slab per blockIdx.y, reduced later
  if (ln < OUT_) {
    float* p = part + (size_t)blockIdx.y * ((size_t)B_ * OUT_);
#pragma unroll
    for (int r = 0; r < 8; ++r) {
      int M = wid * 16 + r + 8 * half;
      p[(size_t)(rb + M) * OUT_ + ln] = oacc[r];
    }
  }
}

__global__ __launch_bounds__(256) void reduce_out_kernel(const float* __restrict__ part,
                                                         float* __restrict__ out) {
  int i = blockIdx.x * 256 + threadIdx.x;
  if (i < B_ * OUT_) {
    float s = 0.f;
#pragma unroll
    for (int p = 0; p < 8; ++p) s += part[(size_t)p * ((size_t)B_ * OUT_) + i];
    out[i] = s;
  }
}

extern "C" void kernel_launch(void* const* d_in, const int* in_sizes, int n_in,
                              void* d_out, int out_size, void* d_ws, size_t ws_size,
                              hipStream_t stream) {
  const float* x     = (const float*)d_in[0];
  const float* loc   = (const float*)d_in[1];
  const float* scale = (const float*)d_in[2];
  const float* w     = (const float*)d_in[3];
  float* out = (float*)d_out;

  float* ws   = (float*)d_ws;
  float* x2   = ws;             // B_ floats
  float* c2   = ws + B_;        // C_ floats
  float* part = ws + B_ + C_;   // 8 * B_ * OUT_ floats (~2.5 MB)

  rowsumsq_kernel<<<B_ / 8, 256, 0, stream>>>(x, x2);
  rowsumsq_kernel<<<C_ / 8, 256, 0, stream>>>(loc, c2);

  dim3 grid(B_ / 128, C_ / 512);   // 64 x 8 = 512 blocks
  rbf_fused_kernel<<<grid, 256, 0, stream>>>(x, loc, scale, w, x2, c2, part);

  reduce_out_kernel<<<(B_ * OUT_ + 255) / 256, 256, 0, stream>>>(part, out);
}
